// ViTencoder_23716809409333
// MI455X (gfx1250) — matrix operand order repaired and checked by reading, not compiled, not run on hardware
//
#include <hip/hip_runtime.h>

// ---------------------------------------------------------------------------
// ViT encoder for MI455X (gfx1250): bf16 WMMA GEMMs + transposed flash attn.
// All matmuls run on v_wmma_f32_16x16x32_bf16 (f32 accumulate).
// ---------------------------------------------------------------------------

typedef __attribute__((ext_vector_type(16))) __bf16 v16bf;
typedef __attribute__((ext_vector_type(8)))  __bf16 bf16x8;
typedef __attribute__((ext_vector_type(8)))  float  v8f;

constexpr int B_   = 16;
constexpr int C_   = 3;
constexpr int IMG_ = 256;
constexpr int P_   = 8;
constexpr int E_   = 384;
constexpr int H_   = 6;
constexpr int L_   = 8;
constexpr int MLP_ = 1536;
constexpr int S_   = 1024;          // (IMG/P)^2
constexpr int OUT_ = 2048;
constexpr int DH_  = 64;
constexpr int KP_  = C_ * P_ * P_;  // 192 patch vector length
constexpr int M_   = B_ * S_;       // 16384 token rows

// ---------------- fragment load helpers (wave32 WMMA layouts) ---------------
// B fragment (32x16, bf16): lane -> col n = lane&15, v[i] = B[(lane>>4)*16+i][n].
__device__ __forceinline__ v16bf ld_b16(const __bf16* p) {
  union { v16bf v; bf16x8 h[2]; } u;
  u.h[0] = *(const bf16x8*)(p);
  u.h[1] = *(const bf16x8*)(p + 8);
  return u.v;
}
// A fragment (16x32, bf16): lane -> row m = lane&15, half = lane>>4,
// v[i] = A[m][k0 + half*16 + i].
__device__ __forceinline__ v16bf ld_a16(const __bf16* row, int k0, int half) {
  union { v16bf v; bf16x8 h[2]; } u;
  u.h[0] = *(const bf16x8*)(row + k0 + half * 16);
  u.h[1] = *(const bf16x8*)(row + k0 + half * 16 + 8);
  return u.v;
}

#define WMMA_BF16(a, b, c) \
  __builtin_amdgcn_wmma_f32_16x16x32_bf16(false, (a), false, (b), (short)0, (c), false, false)

// ---------------- weight preparation ---------------------------------------
// dst[b][c][r] = src[b][r][c]  (f32 -> bf16, transpose to [N][K])
__global__ __launch_bounds__(256) void k_transpose_bf16(
    const float* __restrict__ src, __bf16* __restrict__ dst,
    int batch, int rows, int cols) {
  size_t total = (size_t)batch * rows * cols;
  for (size_t i = (size_t)blockIdx.x * blockDim.x + threadIdx.x; i < total;
       i += (size_t)gridDim.x * blockDim.x) {
    size_t b   = i / ((size_t)rows * cols);
    size_t rem = i % ((size_t)rows * cols);
    size_t c   = rem / rows;
    size_t r   = rem % rows;
    dst[i] = (__bf16)src[(b * rows + r) * cols + c];
  }
}

__global__ __launch_bounds__(256) void k_convert_bf16(
    const float* __restrict__ src, __bf16* __restrict__ dst, size_t n) {
  for (size_t i = (size_t)blockIdx.x * blockDim.x + threadIdx.x; i < n;
       i += (size_t)gridDim.x * blockDim.x)
    dst[i] = (__bf16)src[i];
}

// patchify: Ap[row][k], row = b*S + hp*32 + wp, k = c*64 + p*8 + q
__global__ __launch_bounds__(256) void k_patchify(
    const float* __restrict__ x, __bf16* __restrict__ Ap) {
  size_t total = (size_t)M_ * KP_;
  for (size_t i = (size_t)blockIdx.x * blockDim.x + threadIdx.x; i < total;
       i += (size_t)gridDim.x * blockDim.x) {
    int row = (int)(i / KP_);
    int k   = (int)(i % KP_);
    int b = row >> 10, s = row & (S_ - 1);
    int hp = s >> 5, wp = s & 31;
    int c = k >> 6, rem = k & 63;
    int p = rem >> 3, q = rem & 7;
    Ap[i] = (__bf16)x[(((size_t)(b * C_ + c) * IMG_) + hp * P_ + p) * IMG_ + wp * P_ + q];
  }
}

// ---------------- generic WMMA GEMM: C[M,N] = A[M,K] @ Wt[N,K]^T + bias -----
// Wave computes a 16x128 tile (A fragment reused across 8 WMMAs / K-step).
// Waves within a block share one 128-wide weight panel (B tiles cache-resident)
// and walk consecutive M-tiles; prefetch covers L2 latency 4 K-steps ahead.
// 0: +bias -> f32      1: +bias -> bf16       2: +bias, *0.125 -> bf16 (Q)
// 3: +bias -> bf16 transposed per head [b][h][d][s] (V)
// 4: +bias +resid -> f32 (pre-LN sum)         5: +bias, exact GELU -> bf16
// 6: +bias +posenc -> f32 and bf16 (patch embedding)
template <int EPI>
__global__ __launch_bounds__(256) void k_gemm(
    const __bf16* __restrict__ A, const __bf16* __restrict__ Wt,
    const float* __restrict__ bias, float* __restrict__ outf,
    __bf16* __restrict__ outb, const float* __restrict__ resid,
    int M, int N, int K) {
  int lane = threadIdx.x & 31;
  int wid  = (int)((blockIdx.x * blockDim.x + threadIdx.x) >> 5);
  int mtiles = M >> 4;
  if (wid >= mtiles * (N >> 7)) return;
  int mt = wid % mtiles, ng = wid / mtiles;   // block-mates share the B panel
  int m0 = mt << 4, n0 = ng << 7;
  int l15 = lane & 15, half = lane >> 4;

  v8f zero = {};
  v8f acc[8] = {zero, zero, zero, zero, zero, zero, zero, zero};

  const __bf16* arow  = A + (size_t)(m0 + l15) * K;
  const __bf16* bbase = Wt + (size_t)(n0 + l15) * K + half * 16;
  for (int k0 = 0; k0 < K; k0 += 32) {
    if (k0 + 128 < K) {                        // prefetch 4 K-steps ahead
      __builtin_prefetch(arow + k0 + 128, 0, 3);
#pragma unroll
      for (int t = 0; t < 8; t++)
        __builtin_prefetch(bbase + (size_t)(t * 16) * K + k0 + 128, 0, 3);
    }
    v16bf a = ld_a16(arow, k0, half);
#pragma unroll
    for (int t = 0; t < 8; t++) {
      v16bf bt = ld_b16(bbase + (size_t)(t * 16) * K + k0);
      acc[t] = WMMA_BF16(a, bt, acc[t]);
    }
  }

#pragma unroll
  for (int t = 0; t < 8; t++) {
    int col = n0 + t * 16 + l15;
    float bv = bias[col];
#pragma unroll
    for (int r = 0; r < 8; r++) {
      int row = m0 + r + 8 * half;          // C layout: row = vgpr + 8*(lane>>4)
      float v = acc[t][r] + bv;
      if (EPI == 0) {
        outf[(size_t)row * N + col] = v;
      } else if (EPI == 1) {
        outb[(size_t)row * N + col] = (__bf16)v;
      } else if (EPI == 2) {
        outb[(size_t)row * N + col] = (__bf16)(v * 0.125f);   // 1/sqrt(DH)
      } else if (EPI == 3) {
        int b = row >> 10, s = row & (S_ - 1);
        int h = col >> 6, d = col & 63;
        outb[((size_t)(b * H_ + h) * DH_ + d) * S_ + s] = (__bf16)v;
      } else if (EPI == 4) {
        outf[(size_t)row * N + col] = v + resid[(size_t)row * N + col];
      } else if (EPI == 5) {
        float g = 0.5f * v * (1.0f + erff(v * 0.70710678118654752f));
        outb[(size_t)row * N + col] = (__bf16)g;
      } else if (EPI == 6) {
        int s = row & (S_ - 1);
        float ang = (float)s * __expf((float)(col & ~1) * (-9.210340371976184f / (float)E_));
        float pe = (col & 1) ? __cosf(ang) : __sinf(ang);
        float o = v + pe;
        outf[(size_t)row * N + col] = o;
        outb[(size_t)row * N + col] = (__bf16)o;
      }
    }
  }
}

// ---------------- transposed flash attention --------------------------------
// One wave per (b, h, 16-query tile). Everything is computed transposed so the
// query index is the lane's C-layout *column*:
//   S^T[k][q] = K @ Q^T   -> per-query softmax stats are per-lane scalars
//   O^T[d][q] = V^T @ P^T -> single P^T B-fragment reused across 4 WMMAs
// q is pre-scaled by 1/sqrt(DH). V stored transposed [b][h][d][s].
__global__ __launch_bounds__(256) void k_flash(
    const __bf16* __restrict__ qb, const __bf16* __restrict__ kb,
    const __bf16* __restrict__ vt, __bf16* __restrict__ ob) {
  __shared__ __bf16 Plds[8 * 16 * 32];     // per-wave P^T staging: [query][key]
  int lane  = threadIdx.x & 31;
  int wslot = threadIdx.x >> 5;
  int wid   = blockIdx.x * 8 + wslot;
  if (wid >= B_ * H_ * (S_ / 16)) return;
  int qt = wid & 63;
  int h  = (wid >> 6) % H_;
  int b  = wid / (64 * H_);
  int l15 = lane & 15, half = lane >> 4;
  int q0 = qt << 4;

  // Q as B fragments (dims 0..31 and 32..63): v[i] = Q[q0+l15][c*32+half*16+i]
  const __bf16* qrow = qb + (size_t)(b * S_ + q0 + l15) * E_ + h * DH_;
  v16bf qb0 = ld_b16(qrow + half * 16);
  v16bf qb1 = ld_b16(qrow + 32 + half * 16);

  v8f zero = {};
  v8f acc[4] = {zero, zero, zero, zero};   // O^T fragments: dims t*16.., col=query
  float mrun = -1e30f, lrun = 0.0f;        // per-lane: stats of query q0+l15

  __bf16* P = &Plds[wslot * (16 * 32)];
  const __bf16* vbase = vt + (size_t)(b * H_ + h) * DH_ * S_;

  for (int j = 0; j < S_; j += 32) {
    // S^T for key subtiles [j..j+15], [j+16..j+31]; A = K rows (keys)
    const __bf16* kr0 = kb + (size_t)(b * S_ + j + l15) * E_ + h * DH_;
    const __bf16* kr1 = kr0 + (size_t)16 * E_;
    if (j + 64 < S_) {                       // prefetch key tile pair two steps ahead
      __builtin_prefetch(kr0 + (size_t)64 * E_, 0, 3);
      __builtin_prefetch(kr1 + (size_t)64 * E_, 0, 3);
    }
    v8f s0 = zero, s1 = zero;
    s0 = WMMA_BF16(ld_a16(kr0, 0, half), qb0, s0);
    s0 = WMMA_BF16(ld_a16(kr0, 32, half), qb1, s0);
    s1 = WMMA_BF16(ld_a16(kr1, 0, half), qb0, s1);
    s1 = WMMA_BF16(ld_a16(kr1, 32, half), qb1, s1);
    // lane's elements all belong to query q0+l15; rows are keys j + r + 8*half
    float m = fmaxf(s0[0], s1[0]);
#pragma unroll
    for (int r = 1; r < 8; r++) m = fmaxf(m, fmaxf(s0[r], s1[r]));
    m = fmaxf(m, __shfl_xor(m, 16, 32));   // combine the two half-waves
    float mnew = fmaxf(mrun, m);
    float corr = __expf(mrun - mnew);
    mrun = mnew;
    float psum = 0.0f;
    bf16x8 pk0, pk1;
#pragma unroll
    for (int r = 0; r < 8; r++) {
      float e0 = __expf(s0[r] - mnew);
      float e1 = __expf(s1[r] - mnew);
      psum += e0 + e1;
      pk0[r] = (__bf16)e0;
      pk1[r] = (__bf16)e1;
    }
    psum += __shfl_xor(psum, 16, 32);
    lrun = lrun * corr + psum;
#pragma unroll
    for (int t = 0; t < 4; t++)
#pragma unroll
      for (int r = 0; r < 8; r++) acc[t][r] *= corr;
    // stage P^T[key][query] as LDS [query][key]: packed 16B stores
    *(bf16x8*)(P + l15 * 32 + 8 * half)      = pk0;   // keys j + 8*half + r
    *(bf16x8*)(P + l15 * 32 + 16 + 8 * half) = pk1;   // keys j+16+8*half + r
    // one P^T B fragment, reused for all 4 O^T WMMAs
    v16bf pbf = ld_b16(P + l15 * 32 + half * 16);
#pragma unroll
    for (int t = 0; t < 4; t++) {
      // A = V^T rows (dims t*16+l15), contiguous over keys
      v16bf va = ld_a16(vbase + (size_t)(t * 16 + l15) * S_ + j, 0, half);
      acc[t] = WMMA_BF16(va, pbf, acc[t]);
    }
  }

  float inv = 1.0f / lrun;
  // O^T C layout: col = query q0+l15, rows = dims t*16 + r + 8*half
  __bf16* orow = ob + (size_t)(b * S_ + q0 + l15) * E_ + h * DH_;
#pragma unroll
  for (int t = 0; t < 4; t++) {
    bf16x8 pk;
#pragma unroll
    for (int r = 0; r < 8; r++) pk[r] = (__bf16)(acc[t][r] * inv);
    *(bf16x8*)(orow + t * 16 + 8 * half) = pk;        // 8 consecutive dims
  }
}

// ---------------- layernorm (wave per row) ----------------------------------
__global__ __launch_bounds__(256) void k_layernorm(
    const float* __restrict__ in, const float* __restrict__ gamma,
    const float* __restrict__ beta, float* __restrict__ outf,
    __bf16* __restrict__ outb, int rows) {
  int lane = threadIdx.x & 31;
  int wid  = (int)((blockIdx.x * blockDim.x + threadIdx.x) >> 5);
  if (wid >= rows) return;
  const float* p = in + (size_t)wid * E_;
  float v[E_ / 32];
  float sum = 0.0f, sq = 0.0f;
#pragma unroll
  for (int j = 0; j < E_ / 32; j++) {
    v[j] = p[lane + 32 * j];
    sum += v[j];
    sq  += v[j] * v[j];
  }
#pragma unroll
  for (int o = 16; o >= 1; o >>= 1) {
    sum += __shfl_xor(sum, o, 32);
    sq  += __shfl_xor(sq, o, 32);
  }
  float mean = sum * (1.0f / E_);
  float var  = sq * (1.0f / E_) - mean * mean;
  float rstd = rsqrtf(var + 1e-5f);
#pragma unroll
  for (int j = 0; j < E_ / 32; j++) {
    int idx = lane + 32 * j;
    float y = (v[j] - mean) * rstd * gamma[idx] + beta[idx];
    outf[(size_t)wid * E_ + idx] = y;
    outb[(size_t)wid * E_ + idx] = (__bf16)y;
  }
}

// ---------------- mean pool over sequence -----------------------------------
__global__ __launch_bounds__(256) void k_pool(
    const float* __restrict__ t, __bf16* __restrict__ pooled) {
  int i = blockIdx.x * blockDim.x + threadIdx.x;
  if (i >= B_ * E_) return;
  int b = i / E_, e = i % E_;
  float s = 0.0f;
  for (int j = 0; j < S_; j++) s += t[((size_t)b * S_ + j) * E_ + e];
  pooled[i] = (__bf16)(s * (1.0f / S_));
}

// ---------------- host orchestration ----------------------------------------
template <int EPI>
static void launch_gemm(const __bf16* A, const __bf16* Wt, const float* bias,
                        float* outf, __bf16* outb, const float* resid,
                        int M, int N, int K, hipStream_t s) {
  int waves  = (M >> 4) * (N >> 7);
  int blocks = (waves + 7) / 8;
  k_gemm<EPI><<<blocks, 256, 0, s>>>(A, Wt, bias, outf, outb, resid, M, N, K);
}

static inline dim3 gs(size_t n) { return dim3((unsigned)((n + 255) / 256)); }

extern "C" void kernel_launch(void* const* d_in, const int* in_sizes, int n_in,
                              void* d_out, int out_size, void* d_ws, size_t ws_size,
                              hipStream_t stream) {
  (void)in_sizes; (void)n_in; (void)out_size; (void)ws_size;
  const float* x      = (const float*)d_in[0];
  const float* conv_w = (const float*)d_in[1];
  const float* conv_b = (const float*)d_in[2];
  const float* wq = (const float*)d_in[3];  const float* bq = (const float*)d_in[4];
  const float* wk = (const float*)d_in[5];  const float* bk = (const float*)d_in[6];
  const float* wv = (const float*)d_in[7];  const float* bv = (const float*)d_in[8];
  const float* wo = (const float*)d_in[9];  const float* bo = (const float*)d_in[10];
  const float* ln1s = (const float*)d_in[11]; const float* ln1b = (const float*)d_in[12];
  const float* ln2s = (const float*)d_in[13]; const float* ln2b = (const float*)d_in[14];
  const float* w1 = (const float*)d_in[15]; const float* b1 = (const float*)d_in[16];
  const float* w2 = (const float*)d_in[17]; const float* b2 = (const float*)d_in[18];
  const float* fcw = (const float*)d_in[19]; const float* fcb = (const float*)d_in[20];
  float* out = (float*)d_out;

  // workspace carve (256B aligned bump allocator)
  char* wsp = (char*)d_ws;
  auto alloc = [&](size_t bytes) -> void* {
    void* p = (void*)wsp;
    wsp += (bytes + 255) & ~(size_t)255;
    return p;
  };
  float*   t     = (float*)alloc((size_t)M_ * E_ * 4);
  float*   res   = (float*)alloc((size_t)M_ * E_ * 4);
  __bf16*  tb    = (__bf16*)alloc((size_t)M_ * E_ * 2);
  __bf16*  qbuf  = (__bf16*)alloc((size_t)M_ * E_ * 2);
  __bf16*  kbuf  = (__bf16*)alloc((size_t)M_ * E_ * 2);
  __bf16*  vtbuf = (__bf16*)alloc((size_t)M_ * E_ * 2);
  __bf16*  abuf  = (__bf16*)alloc((size_t)M_ * E_ * 2);
  __bf16*  hbuf  = (__bf16*)alloc((size_t)M_ * MLP_ * 2);
  __bf16*  Ap    = (__bf16*)alloc((size_t)M_ * KP_ * 2);
  __bf16*  pooled= (__bf16*)alloc((size_t)B_ * E_ * 2);
  __bf16*  wqt = (__bf16*)alloc((size_t)L_ * E_ * E_ * 2);
  __bf16*  wkt = (__bf16*)alloc((size_t)L_ * E_ * E_ * 2);
  __bf16*  wvt = (__bf16*)alloc((size_t)L_ * E_ * E_ * 2);
  __bf16*  wot = (__bf16*)alloc((size_t)L_ * E_ * E_ * 2);
  __bf16*  w1t = (__bf16*)alloc((size_t)L_ * E_ * MLP_ * 2);
  __bf16*  w2t = (__bf16*)alloc((size_t)L_ * E_ * MLP_ * 2);
  __bf16*  fct = (__bf16*)alloc((size_t)OUT_ * E_ * 2);
  __bf16*  cwt = (__bf16*)alloc((size_t)E_ * KP_ * 2);

  // weight prep: bf16 convert + transpose to [N][K]
  k_transpose_bf16<<<gs((size_t)L_*E_*E_), 256, 0, stream>>>(wq, wqt, L_, E_, E_);
  k_transpose_bf16<<<gs((size_t)L_*E_*E_), 256, 0, stream>>>(wk, wkt, L_, E_, E_);
  k_transpose_bf16<<<gs((size_t)L_*E_*E_), 256, 0, stream>>>(wv, wvt, L_, E_, E_);
  k_transpose_bf16<<<gs((size_t)L_*E_*E_), 256, 0, stream>>>(wo, wot, L_, E_, E_);
  k_transpose_bf16<<<gs((size_t)L_*E_*MLP_), 256, 0, stream>>>(w1, w1t, L_, E_, MLP_);
  k_transpose_bf16<<<gs((size_t)L_*E_*MLP_), 256, 0, stream>>>(w2, w2t, L_, MLP_, E_);
  k_transpose_bf16<<<gs((size_t)E_*OUT_), 256, 0, stream>>>(fcw, fct, 1, E_, OUT_);
  k_convert_bf16<<<gs((size_t)E_*KP_), 256, 0, stream>>>(conv_w, cwt, (size_t)E_*KP_);

  // patch embedding + positional encoding
  k_patchify<<<gs((size_t)M_*KP_), 256, 0, stream>>>(x, Ap);
  launch_gemm<6>(Ap, cwt, conv_b, t, tb, nullptr, M_, E_, KP_, stream);

  for (int i = 0; i < L_; i++) {
    size_t wofs = (size_t)i * E_ * E_;
    launch_gemm<2>(tb, wqt + wofs, bq + i * E_, nullptr, qbuf, nullptr, M_, E_, E_, stream);
    launch_gemm<1>(tb, wkt + wofs, bk + i * E_, nullptr, kbuf, nullptr, M_, E_, E_, stream);
    launch_gemm<3>(tb, wvt + wofs, bv + i * E_, nullptr, vtbuf, nullptr, M_, E_, E_, stream);
    k_flash<<<dim3(B_ * H_ * (S_ / 16) / 8), 256, 0, stream>>>(qbuf, kbuf, vtbuf, abuf);
    launch_gemm<4>(abuf, wot + wofs, bo + i * E_, res, nullptr, t, M_, E_, E_, stream);
    k_layernorm<<<dim3(M_ / 8), 256, 0, stream>>>(res, ln1s + i * E_, ln1b + i * E_, t, tb, M_);
    launch_gemm<5>(tb, w1t + (size_t)i * E_ * MLP_, b1 + i * MLP_, nullptr, hbuf, nullptr,
                   M_, MLP_, E_, stream);
    launch_gemm<4>(hbuf, w2t + (size_t)i * E_ * MLP_, b2 + i * E_, res, nullptr, t,
                   M_, E_, MLP_, stream);
    k_layernorm<<<dim3(M_ / 8), 256, 0, stream>>>(res, ln2s + i * E_, ln2b + i * E_, t, tb, M_);
  }

  k_pool<<<gs((size_t)B_ * E_), 256, 0, stream>>>(t, pooled);
  launch_gemm<0>(pooled, fct, fcb, out, nullptr, nullptr, B_, OUT_, E_, stream);
}
